// QuantLinearLUT_22497038696508
// MI455X (gfx1250) — matrix operand order repaired and checked by reading, not compiled, not run on hardware
//
#include <hip/hip_runtime.h>
#include <hip/hip_bf16.h>
#include <stdint.h>

typedef __bf16 bf16;
typedef __attribute__((ext_vector_type(16))) __bf16 v16bf;
typedef __attribute__((ext_vector_type(8)))  __bf16 v8bf;
typedef __attribute__((ext_vector_type(8)))  float   v8f;
typedef __attribute__((ext_vector_type(4)))  float   v4f;
typedef __attribute__((ext_vector_type(4)))  int     v4i;

#if __has_builtin(__builtin_amdgcn_global_load_async_to_lds_b128)
#define HAVE_ASYNC_LDS 1
#else
#define HAVE_ASYNC_LDS 0
#endif

typedef __attribute__((address_space(1))) v4i* as1_v4i;
typedef __attribute__((address_space(3))) v4i* as3_v4i;

constexpr int IN_F  = 4096;
constexpr int OUT_F = 4096;
constexpr int TOPX  = 10;
constexpr int M_TOK = 4 * 2048;   // B*S tokens

constexpr int BM = 128, BN = 256, BK = 32;

// ---------------- Phase 1: LUT dequant (4-bit codes -> fp32 W) ----------------
__global__ void k_dequant(const int* __restrict__ qw, const float* __restrict__ lut,
                          float* __restrict__ Wf) {
  int idx = blockIdx.x * blockDim.x + threadIdx.x;     // over (IN_F/8)*OUT_F words
  int n = idx % OUT_F;
  int r = idx / OUT_F;
  if (r >= IN_F / 8) return;
  unsigned w = (unsigned)qw[(size_t)r * OUT_F + n];
  const float* l = lut + (size_t)n * 16;
#pragma unroll
  for (int p = 0; p < 8; ++p) {
    Wf[(size_t)(r * 8 + p) * OUT_F + n] = l[(w >> (4 * p)) & 0xFu];
  }
}

// ---------------- Phase 2: CSR outlier scatter-add ----------------
__global__ void k_csr(const int* __restrict__ rows, const int* __restrict__ cols,
                      const float* __restrict__ vals, float* __restrict__ Wf, int nnz) {
  int k = blockIdx.x * blockDim.x + threadIdx.x;
  if (k >= nnz) return;
  int lo = 0, hi = OUT_F + 1;               // first idx with rows[idx] > k
  while (lo < hi) {
    int mid = (lo + hi) >> 1;
    if (rows[mid] <= k) lo = mid + 1; else hi = mid;
  }
  int row = lo - 1;
  if (row < 0) row = 0;
  if (row >= OUT_F) row = OUT_F - 1;
  atomicAdd(&Wf[(size_t)cols[k] * OUT_F + row], vals[k]);
}

// ---------------- Phase 3: dense outlier columns ----------------
__global__ void k_dense(const float* __restrict__ fr, const int* __restrict__ fri,
                        float* __restrict__ Wf) {
  int idx = blockIdx.x * blockDim.x + threadIdx.x;     // IN_F * TOPX
  if (idx >= IN_F * TOPX) return;
  int t = idx % TOPX;
  int i = idx / TOPX;
  Wf[(size_t)i * OUT_F + fri[t]] += fr[(size_t)i * TOPX + t];   // fri entries distinct
}

// ---------------- Phase 4: fp32 -> bf16, transposed to [N][K] ----------------
__global__ void k_transpose_bf16(const float* __restrict__ Wf, bf16* __restrict__ Wt) {
  int idx = blockIdx.x * blockDim.x + threadIdx.x;     // IN_F*OUT_F
  int k = idx % IN_F;         // consecutive threads -> consecutive k: coalesced stores
  int n = idx / IN_F;
  Wt[(size_t)n * IN_F + k] = (bf16)Wf[(size_t)k * OUT_F + n];
}

// ---------------- Phase 5: bf16 WMMA GEMM, fp32 accumulate, fused bias ----------
// grid = (OUT_F/BN, M_TOK/BM); block = 256 threads = 8 wave32 (2 M x 4 N), wave tile
// 64x64 = 4x4 wmma fragments -> 16 v_wmma per K-step vs 16 ds_load_b128.
__global__ __launch_bounds__(256)
void k_gemm(const float* __restrict__ X, const bf16* __restrict__ Wt,
            const float* __restrict__ bias, float* __restrict__ Y) {
  __shared__ __align__(16) bf16 As[BM][BK];   // [m][k], 8 KB
  __shared__ __align__(16) bf16 Bs[BN][BK];   // [n][k], 16 KB (K-contiguous per column)

  const int bn   = blockIdx.x * BN;
  const int bm   = blockIdx.y * BM;
  const int tid  = threadIdx.x;
  const int wave = tid >> 5;
  const int lane = tid & 31;
  const int wm   = (wave >> 2) * 64;          // wave M offset: {0, 64}
  const int wn   = (wave & 3)  * 64;          // wave N offset: {0,64,128,192}

  const v8f zero = {0.f, 0.f, 0.f, 0.f, 0.f, 0.f, 0.f, 0.f};
  v8f acc[4][4];
#pragma unroll
  for (int f = 0; f < 4; ++f)
#pragma unroll
    for (int j = 0; j < 4; ++j) acc[f][j] = zero;

  for (int k0 = 0; k0 < IN_F; k0 += BK) {
    // ---- stage B tile: 256n x 32k bf16 via async global->LDS DMA (ASYNCcnt) ----
    // chunk id = n*4 + c (16B chunks along K): consecutive tids read consecutive
    // global 16B chunks -> fully coalesced; no VGPR round-trip on the async path.
#pragma unroll
    for (int i = 0; i < 4; ++i) {
      int cid = i * 256 + tid;
      int n = cid >> 2;
      int c = cid & 3;
      const bf16* src = Wt + (size_t)(bn + n) * IN_F + k0 + c * 8;
#if HAVE_ASYNC_LDS
      __builtin_amdgcn_global_load_async_to_lds_b128(
          (as1_v4i)(v4i*)const_cast<bf16*>(src),
          (as3_v4i)(v4i*)&Bs[n][c * 8], 0, 0);
#else
      *(v8bf*)&Bs[n][c * 8] = *(const v8bf*)src;
#endif
    }
    // ---- stage A tile: 128m x 32k, fp32 -> bf16 (cvt co-executes with XDL) ----
    {
      int m  = tid >> 1;
      int kk = (tid & 1) * 16;
      const float* src = X + (size_t)(bm + m) * IN_F + k0 + kk;
      v4f x0 = *(const v4f*)(src + 0);
      v4f x1 = *(const v4f*)(src + 4);
      v4f x2 = *(const v4f*)(src + 8);
      v4f x3 = *(const v4f*)(src + 12);
      v8bf o0, o1;
#pragma unroll
      for (int i = 0; i < 4; ++i) {
        o0[i]     = (bf16)x0[i];
        o0[4 + i] = (bf16)x1[i];
        o1[i]     = (bf16)x2[i];
        o1[4 + i] = (bf16)x3[i];
      }
      *(v8bf*)&As[m][kk]     = o0;
      *(v8bf*)&As[m][kk + 8] = o1;
      if (k0 + BK < IN_F) __builtin_prefetch(src + BK, 0, 1);   // next K-slab
    }
#if HAVE_ASYNC_LDS
#if __has_builtin(__builtin_amdgcn_s_wait_asynccnt)
    __builtin_amdgcn_s_wait_asynccnt(0);
#else
    asm volatile("s_wait_asynccnt 0x0" ::: "memory");
#endif
#endif
    __syncthreads();

    // ---- fragments per ISA wave32 layouts (§7.12.2) ----
    // A 16x32 bf16: lanes 0-15 row m=lane, K {0..7,16..23}; lanes 16-31 K {8..15,24..31}
    v16bf afrag[4];
#pragma unroll
    for (int f = 0; f < 4; ++f) {
      int ma = wm + f * 16 + (lane & 15);
      int ka = (lane >> 4) * 8;
      v8bf alo = *(const v8bf*)&As[ma][ka];
      v8bf ahi = *(const v8bf*)&As[ma][ka + 16];
      afrag[f] = __builtin_shufflevector(alo, ahi, 0, 1, 2, 3, 4, 5, 6, 7,
                                                   8, 9, 10, 11, 12, 13, 14, 15);
    }
    // B 32x16 bf16: lane col n=lane%16, lanes 0-15 K {0..7,16..23}; lanes 16-31 K {8..15,24..31}
    v16bf bfrag[4];
#pragma unroll
    for (int j = 0; j < 4; ++j) {
      int nb = wn + j * 16 + (lane & 15);
      int kb = (lane >> 4) * 8;
      v8bf blo = *(const v8bf*)&Bs[nb][kb];
      v8bf bhi = *(const v8bf*)&Bs[nb][kb + 16];
      bfrag[j] = __builtin_shufflevector(blo, bhi, 0, 1, 2, 3, 4, 5, 6, 7,
                                                   8, 9, 10, 11, 12, 13, 14, 15);
    }

#pragma unroll
    for (int f = 0; f < 4; ++f)
#pragma unroll
      for (int j = 0; j < 4; ++j)
        acc[f][j] = __builtin_amdgcn_wmma_f32_16x16x32_bf16(
            false, afrag[f], false, bfrag[j], (short)0, acc[f][j], false, false);

    __syncthreads();
  }

  // ---- epilogue: C 16x16 f32 layout -> global, fused bias ----
  // lanes 0-15: N=lane, M=r; lanes 16-31: N=lane-16, M=8+r
#pragma unroll
  for (int f = 0; f < 4; ++f) {
#pragma unroll
    for (int j = 0; j < 4; ++j) {
      int nc = bn + wn + j * 16 + (lane & 15);
      int mc = bm + wm + f * 16 + ((lane >> 4) * 8);
      float bv = bias[nc];
#pragma unroll
      for (int r = 0; r < 8; ++r)
        Y[(size_t)(mc + r) * OUT_F + nc] = acc[f][j][r] + bv;
    }
  }
}

extern "C" void kernel_launch(void* const* d_in, const int* in_sizes, int n_in,
                              void* d_out, int out_size, void* d_ws, size_t ws_size,
                              hipStream_t stream) {
  const float* x     = (const float*)d_in[0];
  const int*   qw    = (const int*)  d_in[1];
  const float* lut   = (const float*)d_in[2];
  const float* bias  = (const float*)d_in[3];
  const int*   rows  = (const int*)  d_in[4];
  const int*   cols  = (const int*)  d_in[5];
  const float* vals  = (const float*)d_in[6];
  const float* frow  = (const float*)d_in[7];
  const int*   fri   = (const int*)  d_in[8];
  const int    nnz   = in_sizes[6];

  float* Wf = (float*)d_ws;                                        // 64 MB fp32 W
  bf16*  Wt = (bf16*)((char*)d_ws + (size_t)IN_F * OUT_F * 4);     // +32 MB bf16 W^T

  // Phase 1: dequant
  {
    int total = (IN_F / 8) * OUT_F;
    k_dequant<<<(total + 255) / 256, 256, 0, stream>>>(qw, lut, Wf);
  }
  // Phase 2: CSR outliers
  k_csr<<<(nnz + 255) / 256, 256, 0, stream>>>(rows, cols, vals, Wf, nnz);
  // Phase 3: dense outlier columns
  k_dense<<<(IN_F * TOPX + 255) / 256, 256, 0, stream>>>(frow, fri, Wf);
  // Phase 4: convert + transpose -> bf16 [N][K] (32 MB, fully L2-resident)
  {
    int total = IN_F * OUT_F;
    k_transpose_bf16<<<(total + 255) / 256, 256, 0, stream>>>(Wf, Wt);
  }
  // Phase 5: WMMA GEMM with fused bias
  {
    dim3 grid(OUT_F / BN, M_TOK / BM);
    k_gemm<<<grid, 256, 0, stream>>>(x, Wt, bias, (float*)d_out);
  }
}
